// SelfAttentionV2_64871186039337
// MI455X (gfx1250) — matrix operand order repaired and checked by reading, not compiled, not run on hardware
//
#include <hip/hip_runtime.h>
#include <hip/hip_bf16.h>

// ---------------------------------------------------------------------------
// Causal self-attention, S=4096, d=1024, fp32 I/O, MI455X (gfx1250).
// bf16 WMMA for all GEMMs; A/B tiles staged with GLOBAL_LOAD_ASYNC_TO_LDS_B128
// (ASYNCcnt-tracked DMA, double-buffered); ds_load_b128 fragment reads;
// L2-resident materialized scores; causal tile-skip + K-trim.
// ---------------------------------------------------------------------------

typedef __attribute__((ext_vector_type(16))) __bf16 v16bf;
typedef __attribute__((ext_vector_type(8)))  float  v8f;

union FragU { uint4 u[2]; v16bf v; };

__device__ __forceinline__ unsigned short f32_to_bf16(float f) {
  union { float f; unsigned int u; } c; c.f = f;
  unsigned int r = c.u + 0x7FFFu + ((c.u >> 16) & 1u);  // round-to-nearest-even
  return (unsigned short)(r >> 16);
}

__global__ void cvt_f32_to_bf16(const float* __restrict__ src,
                                unsigned short* __restrict__ dst, int n) {
  int i = (blockIdx.x * blockDim.x + threadIdx.x) * 4;
  if (i < n) {
    float4 f = *(const float4*)(src + i);
    ushort4 o;
    o.x = f32_to_bf16(f.x); o.y = f32_to_bf16(f.y);
    o.z = f32_to_bf16(f.z); o.w = f32_to_bf16(f.w);
    *(ushort4*)(dst + i) = o;
  }
}

#define TILE_M 128
#define TILE_N 128
#define BK     32
#define PADR   40   // LDS row pitch in bf16 elems (80B): 16B-aligned, bank-conflict-free

// mode 0: C bf16 row-major            (Q, K projections)
// mode 1: C bf16 transposed store     (V projection -> V^T)
// mode 2: C f32, *scale, causal mask  (scores; fully-masked tiles fast-pathed)
// mode 3: C f32, causal K-trim        (O = P @ V^T; kEnd = m0 + TILE_M)
__global__ __launch_bounds__(256) void wmma_gemm_kernel(
    const unsigned short* __restrict__ A, int lda,
    const unsigned short* __restrict__ B, int ldb,
    void* __restrict__ Cv, int ldc,
    int K, int mode, float scale)
{
  __shared__ unsigned short ldsA[2][TILE_M * PADR];
  __shared__ unsigned short ldsB[2][TILE_N * PADR];

  const int tid   = threadIdx.x;
  const int lane  = tid & 31;
  const int wave  = tid >> 5;
  const int wm    = wave & 3;            // 4 waves along M (32 rows each)
  const int wn    = wave >> 2;           // 2 waves along N (64 cols each)
  const int m0    = blockIdx.y * TILE_M;
  const int n0    = blockIdx.x * TILE_N;
  const int mloc  = wm * 32;
  const int nloc  = wn * 64;
  const int crow  = (lane >> 4) << 3;    // C layout: M = vgpr + 8*(lane>=16)
  const int ccol  = lane & 15;           //           N = lane & 15

  // Scores: whole block tile strictly above the diagonal -> mask value, no GEMM.
  if (mode == 2 && n0 >= m0 + TILE_M) {
    float* C = (float*)Cv;
    for (int i = 0; i < 2; ++i)
      for (int j = 0; j < 4; ++j)
        for (int r = 0; r < 8; ++r)
          C[(size_t)(m0 + mloc + 16 * i + crow + r) * ldc +
            (n0 + nloc + 16 * j + ccol)] = -3.0e38f;
    return;
  }

  int kEnd = K;
  if (mode == 3) { int t = m0 + TILE_M; kEnd = (t < K) ? t : K; }  // P==0 past diag

  // --- Per-thread staging geometry: 512 b128 chunks per matrix slice, 256
  // threads -> each thread owns 2 A-chunks and 2 B-chunks. Precompute once.
  const int c0 = tid, c1 = tid + 256;
  const int r0 = c0 >> 2, h0 = (c0 & 3) << 3;   // chunk row / elem offset
  const int r1 = c1 >> 2, h1 = (c1 & 3) << 3;
  unsigned gA0 = (unsigned)(((size_t)(m0 + r0) * lda + h0) * 2);   // byte offsets
  unsigned gA1 = (unsigned)(((size_t)(m0 + r1) * lda + h1) * 2);
  unsigned gB0 = (unsigned)(((size_t)(n0 + r0) * ldb + h0) * 2);
  unsigned gB1 = (unsigned)(((size_t)(n0 + r1) * ldb + h1) * 2);
  const unsigned lA0 = (unsigned)(uintptr_t)(&ldsA[0][r0 * PADR + h0]);
  const unsigned lA1 = (unsigned)(uintptr_t)(&ldsA[0][r1 * PADR + h1]);
  const unsigned lB0 = (unsigned)(uintptr_t)(&ldsB[0][r0 * PADR + h0]);
  const unsigned lB1 = (unsigned)(uintptr_t)(&ldsB[0][r1 * PADR + h1]);
  const unsigned strA = TILE_M * PADR * 2;      // buffer 1 byte offset (A)
  const unsigned strB = TILE_N * PADR * 2;      // buffer 1 byte offset (B)

  // Async DMA one 128x32 A slice + 128x32 B slice into LDS buffer `buf`.
  auto stage_async = [&](int buf, unsigned kbyte) {
    const unsigned sa = buf ? strA : 0u;
    const unsigned sb = buf ? strB : 0u;
    asm volatile("global_load_async_to_lds_b128 %0, %1, %2"
                 :: "v"(lA0 + sa), "v"(gA0 + kbyte), "s"(A) : "memory");
    asm volatile("global_load_async_to_lds_b128 %0, %1, %2"
                 :: "v"(lA1 + sa), "v"(gA1 + kbyte), "s"(A) : "memory");
    asm volatile("global_load_async_to_lds_b128 %0, %1, %2"
                 :: "v"(lB0 + sb), "v"(gB0 + kbyte), "s"(B) : "memory");
    asm volatile("global_load_async_to_lds_b128 %0, %1, %2"
                 :: "v"(lB1 + sb), "v"(gB1 + kbyte), "s"(B) : "memory");
  };

  auto a_frag = [&](const unsigned short* base, int rowbase) -> v16bf {
    const int row  = rowbase + (lane & 15);
    const int koff = (lane >> 4) << 3;                 // 0 or 8
    FragU f;
    f.u[0] = *(const uint4*)(base + row * PADR + koff);        // K koff..+7
    f.u[1] = *(const uint4*)(base + row * PADR + koff + 16);   // K koff+16..+23
    return f.v;
  };
  auto b_frag = [&](const unsigned short* base, int rowbase) -> v16bf {
    const int row = rowbase + (lane & 15);
    const int kb  = (lane >> 4) << 3;                  // 0 or 8
    FragU f;
    f.u[0] = *(const uint4*)(base + row * PADR + kb);
    f.u[1] = *(const uint4*)(base + row * PADR + kb + 16);
    return f.v;
  };

  const v8f vzero = {0.f, 0.f, 0.f, 0.f, 0.f, 0.f, 0.f, 0.f};
  v8f acc[2][4] = {{vzero, vzero, vzero, vzero}, {vzero, vzero, vzero, vzero}};

  int buf = 0;
  stage_async(0, 0u);

  for (int k0 = 0; k0 < kEnd; k0 += BK) {
    // Wait for our own DMAs into `buf` (next-buffer DMAs not yet issued),
    // then barrier so every wave's slice is visible.
    asm volatile("s_wait_asynccnt 0x0" ::: "memory");
    __syncthreads();

    const int knext = k0 + BK;
    if (knext < kEnd) stage_async(buf ^ 1, (unsigned)(knext * 2));

    if (k0 + 2 * BK < kEnd) {                // uniform guard: prefetch 2 slices out
      const int row = tid >> 1, ch = (tid & 1) << 4;
      __builtin_prefetch(A + (size_t)(m0 + row) * lda + k0 + 2 * BK + ch, 0, 1);
      __builtin_prefetch(B + (size_t)(n0 + row) * ldb + k0 + 2 * BK + ch, 0, 1);
    }

    const unsigned short* pA = ldsA[buf];
    const unsigned short* pB = ldsB[buf];
    v16bf a0 = a_frag(pA, mloc);
    v16bf a1 = a_frag(pA, mloc + 16);
#pragma unroll
    for (int j = 0; j < 4; ++j) {
      v16bf b = b_frag(pB, nloc + 16 * j);
      acc[0][j] = __builtin_amdgcn_wmma_f32_16x16x32_bf16(false, a0, false, b, (short)0, acc[0][j], false, false);
      acc[1][j] = __builtin_amdgcn_wmma_f32_16x16x32_bf16(false, a1, false, b, (short)0, acc[1][j], false, false);
    }
    buf ^= 1;
  }

  for (int i = 0; i < 2; ++i)
    for (int j = 0; j < 4; ++j)
      for (int r = 0; r < 8; ++r) {
        const int gm = m0 + mloc + 16 * i + crow + r;
        const int gn = n0 + nloc + 16 * j + ccol;
        float v = acc[i][j][r];
        if (mode == 2) {
          v *= scale;
          if (gn > gm) v = -3.0e38f;
          ((float*)Cv)[(size_t)gm * ldc + gn] = v;
        } else if (mode == 3) {
          ((float*)Cv)[(size_t)gm * ldc + gn] = v;
        } else if (mode == 1) {
          ((unsigned short*)Cv)[(size_t)gn * ldc + gm] = f32_to_bf16(v);
        } else {
          ((unsigned short*)Cv)[(size_t)gm * ldc + gn] = f32_to_bf16(v);
        }
      }
}

// One 256-thread block per row: 16 f32 values/thread in registers,
// max + sum reductions in LDS, in-place rewrite of the row as bf16 P.
__global__ __launch_bounds__(256) void softmax_row_kernel(float* __restrict__ S, int ld) {
  const int row = blockIdx.x;
  const int tid = threadIdx.x;
  float* rowp = S + (size_t)row * ld;
  unsigned short* prow = (unsigned short*)rowp;   // bf16 P overwrites row start

  float vals[16];
  float m = -3.4e38f;
  for (int t = 0; t < 16; ++t) {
    vals[t] = rowp[tid + t * 256];
    m = fmaxf(m, vals[t]);
  }

  __shared__ float red[256];
  red[tid] = m;
  __syncthreads();
  for (int s = 128; s > 0; s >>= 1) {
    if (tid < s) red[tid] = fmaxf(red[tid], red[tid + s]);
    __syncthreads();
  }
  const float rmax = red[0];
  __syncthreads();

  float lsum = 0.f;
  for (int t = 0; t < 16; ++t) {
    vals[t] = __expf(vals[t] - rmax);   // masked (-3e38) entries -> 0
    lsum += vals[t];
  }
  red[tid] = lsum;
  __syncthreads();
  for (int s = 128; s > 0; s >>= 1) {
    if (tid < s) red[tid] += red[tid + s];
    __syncthreads();
  }
  const float inv = 1.0f / red[0];
  __syncthreads();   // all reads of the f32 row done before bf16 writes

  for (int t = 0; t < 16; ++t)
    prow[tid + t * 256] = f32_to_bf16(vals[t] * inv);
}

extern "C" void kernel_launch(void* const* d_in, const int* in_sizes, int n_in,
                              void* d_out, int out_size, void* d_ws, size_t ws_size,
                              hipStream_t stream) {
  (void)in_sizes; (void)n_in; (void)out_size; (void)ws_size;
  const float* x  = (const float*)d_in[0];
  const float* Wq = (const float*)d_in[1];
  const float* Wk = (const float*)d_in[2];
  const float* Wv = (const float*)d_in[3];

  const int S = 4096, D = 1024;

  // Workspace layout (bytes): ~102 MiB total, fully L2-resident on MI455X.
  char* ws = (char*)d_ws;
  unsigned short* Xb  = (unsigned short*)(ws);                          // 8 MiB  [S,D] bf16
  unsigned short* Wqb = (unsigned short*)(ws + ( 8ull << 20));          // 2 MiB
  unsigned short* Wkb = (unsigned short*)(ws + (10ull << 20));          // 2 MiB
  unsigned short* Wvb = (unsigned short*)(ws + (12ull << 20));          // 2 MiB
  unsigned short* Qb  = (unsigned short*)(ws + (14ull << 20));          // 8 MiB  [S,D]
  unsigned short* Kb  = (unsigned short*)(ws + (22ull << 20));          // 8 MiB  [S,D]
  unsigned short* Vtb = (unsigned short*)(ws + (30ull << 20));          // 8 MiB  [D,S] (V^T)
  float*          Sc  = (float*)        (ws + (38ull << 20));           // 64 MiB [S,S] scores / P(bf16, in place)

  cvt_f32_to_bf16<<<(S * D) / 1024, 256, 0, stream>>>(x,  Xb,  S * D);
  cvt_f32_to_bf16<<<(D * D) / 1024, 256, 0, stream>>>(Wq, Wqb, D * D);
  cvt_f32_to_bf16<<<(D * D) / 1024, 256, 0, stream>>>(Wk, Wkb, D * D);
  cvt_f32_to_bf16<<<(D * D) / 1024, 256, 0, stream>>>(Wv, Wvb, D * D);

  dim3 blk(256);

  // Q = X @ Wq^T, K = X @ Wk^T (bf16 out); V^T stored transposed (mode 1).
  dim3 gProj(D / TILE_N, S / TILE_M);
  wmma_gemm_kernel<<<gProj, blk, 0, stream>>>(Xb, D, Wqb, D, Qb,  D, D, 0, 1.0f);
  wmma_gemm_kernel<<<gProj, blk, 0, stream>>>(Xb, D, Wkb, D, Kb,  D, D, 0, 1.0f);
  wmma_gemm_kernel<<<gProj, blk, 0, stream>>>(Xb, D, Wvb, D, Vtb, S, D, 1, 1.0f);

  // Scores = (1/sqrt(D)) * Q @ K^T with causal mask (f32), masked tiles skipped.
  dim3 gScore(S / TILE_N, S / TILE_M);
  wmma_gemm_kernel<<<gScore, blk, 0, stream>>>(Qb, D, Kb, D, Sc, S, D, 2, 0.03125f);

  // Row softmax -> P (bf16, in place over the score rows; row stride 2*S ushorts).
  softmax_row_kernel<<<S, 256, 0, stream>>>(Sc, S);

  // O = P @ (V^T)^T  (f32 out to d_out), K trimmed to the causal extent.
  dim3 gOut(D / TILE_N, S / TILE_M);
  wmma_gemm_kernel<<<gOut, blk, 0, stream>>>((const unsigned short*)Sc, 2 * S,
                                             Vtb, S, d_out, D, S, 3, 1.0f);
}
